// MultiScaleRetention_55009941127827
// MI455X (gfx1250) — matrix operand order repaired and checked by reading, not compiled, not run on hardware
//
#include <hip/hip_runtime.h>
#include <hip/hip_bf16.h>
#include <math.h>

// ---------------------------------------------------------------------------
// MultiScaleRetention for MI455X (gfx1250, wave32, WMMA + async global->LDS).
//   pack:  x, Wq, Wv, Wg, Wo  -> f16 copies (one-time, halves tile bandwidth)
//   q = x@Wq.T ; v = x@Wv.T ; g = x@Wg.T   (WMMA f16 GEMM, f32 accum,
//                                           double-buffered ASYNC LDS fill)
//   s_t = sigmoid(beta_h)*s_{t-1} + v_t     (chunked parallel scan)
//   y = LayerNorm(q*s) * silu(g) -> f16     (fused row kernel)
//   out = y @ Wo.T                           (WMMA GEMM)
// ---------------------------------------------------------------------------

typedef __attribute__((ext_vector_type(16))) _Float16 h16;
typedef __attribute__((ext_vector_type(8)))  _Float16 h8;
typedef __attribute__((ext_vector_type(4)))  _Float16 h4;
typedef __attribute__((ext_vector_type(8)))  float    v8f;
typedef int v4i_g __attribute__((vector_size(16)));   // matches builtin param type

#define GEMM_BM 128
#define GEMM_BN 128
#define GEMM_BK 32
#define LDSL    40   // halves per LDS row: 32 + 8 pad; row pitch 80B (16B-aligned)

#if __has_builtin(__builtin_amdgcn_global_load_async_to_lds_b128)
#define MSR_HAS_ASYNC 1
#else
#define MSR_HAS_ASYNC 0
#endif

// 16-byte global -> LDS copy (async DMA when available).
__device__ __forceinline__ void msr_cp16(const _Float16* gsrc, _Float16* ldst) {
#if MSR_HAS_ASYNC
    __builtin_amdgcn_global_load_async_to_lds_b128(
        (__attribute__((address_space(1))) v4i_g*)gsrc,
        (__attribute__((address_space(3))) v4i_g*)ldst,
        0, 0);
#else
    *(h8*)ldst = *(const h8*)gsrc;
#endif
}

template <int N>
__device__ __forceinline__ void msr_wait_async() {
#if MSR_HAS_ASYNC
#if __has_builtin(__builtin_amdgcn_s_wait_asynccnt)
    __builtin_amdgcn_s_wait_asynccnt(N);
#else
    asm volatile("s_wait_asynccnt %0" :: "n"(N) : "memory");
#endif
#endif
}

// ---------------------------------------------------------------------------
// fp32 -> f16 packing (one float4 -> h4 per thread)
// ---------------------------------------------------------------------------
__global__ __launch_bounds__(256)
void msr_pack_f16(const float* __restrict__ src, _Float16* __restrict__ dst, int n4) {
    const int i = blockIdx.x * blockDim.x + threadIdx.x;
    if (i < n4) {
        const float4 f = ((const float4*)src)[i];
        h4 h = { (_Float16)f.x, (_Float16)f.y, (_Float16)f.z, (_Float16)f.w };
        ((h4*)dst)[i] = h;
    }
}

// ---------------------------------------------------------------------------
// C[M,N] = A[M,K] @ W[N,K]^T ; A,W f16 row-major (K contiguous), C f32.
// 256 threads = 8 waves (4 along M x 2 along N); 128x128x32 tiles;
// double-buffered async global->LDS fill overlapped with WMMA.
// ---------------------------------------------------------------------------
__global__ __launch_bounds__(256)
void msr_gemm_f16(const _Float16* __restrict__ A, const _Float16* __restrict__ W,
                  float* __restrict__ C, int M, int N, int K) {
    __shared__ _Float16 sA[2][GEMM_BM * LDSL];
    __shared__ _Float16 sB[2][GEMM_BN * LDSL];

    const int tid   = threadIdx.x;
    const int wave  = tid >> 5;
    const int lane  = tid & 31;
    const int waveM = wave & 3;      // 0..3 : 32 rows each
    const int waveN = wave >> 2;     // 0..1 : 64 cols each
    const int bm0   = blockIdx.x * GEMM_BM;
    const int bn0   = blockIdx.y * GEMM_BN;

    v8f acc[2][4];
#pragma unroll
    for (int i = 0; i < 2; ++i)
#pragma unroll
        for (int j = 0; j < 4; ++j) acc[i][j] = (v8f)0.0f;

    // per-lane fragment addressing (contiguous k = 16*(lane/16) + s for A and B, 16-bit operands, wave32)
    const int mlane = lane & 15;
    const int aHalf = (lane < 16) ? 0 : 16;   // A: K range  {aHalf..aHalf+15}
    const int bKofs = (lane < 16) ? 0 : 16;   // B: K range  {bKofs..bKofs+15}

    // tile copy: 128x32 halves = 512 chunks of 16B per tile; 2 chunks/thread/tile
    const int nk = K / GEMM_BK;

#define MSR_ISSUE(buf, k0)                                                        \
    do {                                                                          \
        _Pragma("unroll")                                                         \
        for (int p = 0; p < 2; ++p) {                                             \
            const int id  = tid + p * 256;                                        \
            const int row = id >> 2;                                              \
            const int col = (id & 3) * 8;                                         \
            msr_cp16(A + (size_t)(bm0 + row) * K + (k0) + col,                    \
                     &sA[buf][row * LDSL + col]);                                 \
            msr_cp16(W + (size_t)(bn0 + row) * K + (k0) + col,                    \
                     &sB[buf][row * LDSL + col]);                                 \
        }                                                                         \
    } while (0)

    MSR_ISSUE(0, 0);                       // stage 0 in flight (4 async ops)

    for (int ks = 0; ks < nk; ++ks) {
        const int cur = ks & 1;
        if (ks + 1 < nk) {
            __syncthreads();               // everyone done computing on buf cur^1
            MSR_ISSUE(cur ^ 1, (ks + 1) * GEMM_BK);
            msr_wait_async<4>();           // stage ks landed (newest 4 still flying)
        } else {
            msr_wait_async<0>();           // drain
        }
        __syncthreads();                   // stage ks visible to all waves

        // ---- load fragments from LDS buffer `cur`
        h16 afrag[2];
#pragma unroll
        for (int i = 0; i < 2; ++i) {
            const int row = waveM * 32 + i * 16 + mlane;
            union { h16 v; h8 h[2]; } u;
            u.h[0] = *(const h8*)&sA[cur][row * LDSL + aHalf];
            u.h[1] = *(const h8*)&sA[cur][row * LDSL + aHalf + 8];
            afrag[i] = u.v;
        }
        h16 bfrag[4];
#pragma unroll
        for (int j = 0; j < 4; ++j) {
            const int col = waveN * 64 + j * 16 + mlane;   // n index = row of W tile
            union { h16 v; h8 h[2]; } u;
            u.h[0] = *(const h8*)&sB[cur][col * LDSL + bKofs];
            u.h[1] = *(const h8*)&sB[cur][col * LDSL + bKofs + 8];
            bfrag[j] = u.v;
        }

        // ---- 8 WMMAs per wave per K-step
#pragma unroll
        for (int i = 0; i < 2; ++i)
#pragma unroll
            for (int j = 0; j < 4; ++j)
                acc[i][j] = __builtin_amdgcn_wmma_f32_16x16x32_f16(
                    false, afrag[i], false, bfrag[j],
                    (short)0, acc[i][j], false, false);
    }
#undef MSR_ISSUE

    // ---- epilogue: VGPR r holds M=r(+8 for upper lanes), N=lane&15
    const int mbase = (lane < 16) ? 0 : 8;
#pragma unroll
    for (int i = 0; i < 2; ++i)
#pragma unroll
        for (int j = 0; j < 4; ++j) {
            const int row0 = bm0 + waveM * 32 + i * 16 + mbase;
            const int col  = bn0 + waveN * 64 + j * 16 + mlane;
#pragma unroll
            for (int r = 0; r < 8; ++r)
                C[(size_t)(row0 + r) * N + col] = acc[i][j][r];
        }
}

// ---------------------------------------------------------------------------
// Retention scan: s_t = lam*s_{t-1} + v_t over L, chunked for parallelism.
// ---------------------------------------------------------------------------
#define MSR_B      4
#define MSR_L      4096
#define MSR_D      1024
#define MSR_NCHUNK 16
#define MSR_CLEN   256
#define MSR_CH     (MSR_B * MSR_D)   // 4096 channels

__device__ __forceinline__ float msr_sigmoid(float x) {
    return 1.0f / (1.0f + __expf(-x));
}

__global__ __launch_bounds__(256)
void msr_scan_local(const float* __restrict__ v, const float* __restrict__ beta,
                    float* __restrict__ carry) {
    const int g     = blockIdx.x * blockDim.x + threadIdx.x; // 65536 threads
    const int c     = g & (MSR_D - 1);
    const int rest  = g >> 10;
    const int b     = rest & (MSR_B - 1);
    const int chunk = rest >> 2;
    const float lam = msr_sigmoid(beta[c >> 6]);

    const float* vp = v + ((size_t)b * MSR_L + (size_t)chunk * MSR_CLEN) * MSR_D + c;
    float s = 0.0f;
    for (int i = 0; i < MSR_CLEN; ++i)
        s = fmaf(lam, s, vp[(size_t)i * MSR_D]);
    carry[(size_t)chunk * MSR_CH + b * MSR_D + c] = s;
}

__global__ __launch_bounds__(256)
void msr_scan_chunks(const float* __restrict__ carry, const float* __restrict__ beta,
                     float* __restrict__ sinit) {
    const int g = blockIdx.x * blockDim.x + threadIdx.x;     // 4096 threads
    const int c = g & (MSR_D - 1);
    const int b = g >> 10;
    const float lam = msr_sigmoid(beta[c >> 6]);
    float lamP = lam;                      // lam^256 via 8 squarings
#pragma unroll
    for (int i = 0; i < 8; ++i) lamP *= lamP;

    float s = 0.0f;
    for (int ch = 0; ch < MSR_NCHUNK; ++ch) {
        const size_t idx = (size_t)ch * MSR_CH + b * MSR_D + c;
        sinit[idx] = s;
        s = fmaf(lamP, s, carry[idx]);
    }
}

__global__ __launch_bounds__(256)
void msr_scan_apply(const float* __restrict__ v, float* __restrict__ q_io,
                    const float* __restrict__ beta, const float* __restrict__ sinit) {
    const int g     = blockIdx.x * blockDim.x + threadIdx.x;
    const int c     = g & (MSR_D - 1);
    const int rest  = g >> 10;
    const int b     = rest & (MSR_B - 1);
    const int chunk = rest >> 2;
    const float lam = msr_sigmoid(beta[c >> 6]);

    const size_t base = ((size_t)b * MSR_L + (size_t)chunk * MSR_CLEN) * MSR_D + c;
    const float* vp = v + base;
    float*       qp = q_io + base;
    float s = sinit[(size_t)chunk * MSR_CH + b * MSR_D + c];
    for (int i = 0; i < MSR_CLEN; ++i) {
        const size_t o = (size_t)i * MSR_D;
        s = fmaf(lam, s, vp[o]);
        qp[o] = qp[o] * s;
    }
}

// ---------------------------------------------------------------------------
// y2 = LayerNorm(y)*silu(g) emitted directly as f16 for the final WMMA GEMM.
// One 256-thread block per (b,l) row of 1024. wave32 shuffle reduction.
// ---------------------------------------------------------------------------
__global__ __launch_bounds__(256)
void msr_ln_gate(const float* __restrict__ y, const float* __restrict__ g,
                 const float* __restrict__ gamma, const float* __restrict__ lbeta,
                 _Float16* __restrict__ y2) {
    const size_t row = blockIdx.x;
    const int tid  = threadIdx.x;
    const int lane = tid & 31;
    const int wave = tid >> 5;

    const float4 yv = *(const float4*)(y + row * MSR_D + tid * 4);
    float s1 = yv.x + yv.y + yv.z + yv.w;
    float s2 = yv.x * yv.x + yv.y * yv.y + yv.z * yv.z + yv.w * yv.w;
#pragma unroll
    for (int off = 16; off > 0; off >>= 1) {
        s1 += __shfl_xor(s1, off, 32);
        s2 += __shfl_xor(s2, off, 32);
    }
    __shared__ float r1[8], r2[8];
    if (lane == 0) { r1[wave] = s1; r2[wave] = s2; }
    __syncthreads();
    float t1 = 0.0f, t2 = 0.0f;
#pragma unroll
    for (int i = 0; i < 8; ++i) { t1 += r1[i]; t2 += r2[i]; }

    const float mu  = t1 * (1.0f / MSR_D);
    const float var = t2 * (1.0f / MSR_D) - mu * mu;
    const float rs  = rsqrtf(var + 1e-5f);

    const float4 gm = *(const float4*)(gamma + tid * 4);
    const float4 gb = *(const float4*)(lbeta + tid * 4);
    const float4 gv = *(const float4*)(g + row * MSR_D + tid * 4);

    const float ox = ((yv.x - mu) * rs * gm.x + gb.x) * (gv.x * msr_sigmoid(gv.x));
    const float oy = ((yv.y - mu) * rs * gm.y + gb.y) * (gv.y * msr_sigmoid(gv.y));
    const float oz = ((yv.z - mu) * rs * gm.z + gb.z) * (gv.z * msr_sigmoid(gv.z));
    const float ow = ((yv.w - mu) * rs * gm.w + gb.w) * (gv.w * msr_sigmoid(gv.w));
    h4 out = { (_Float16)ox, (_Float16)oy, (_Float16)oz, (_Float16)ow };
    *(h4*)(y2 + row * MSR_D + tid * 4) = out;
}

// ---------------------------------------------------------------------------
extern "C" void kernel_launch(void* const* d_in, const int* in_sizes, int n_in,
                              void* d_out, int out_size, void* d_ws, size_t ws_size,
                              hipStream_t stream) {
    const float* x     = (const float*)d_in[0];
    const float* Wq    = (const float*)d_in[1];
    const float* Wv    = (const float*)d_in[2];
    const float* Wg    = (const float*)d_in[3];
    const float* Wo    = (const float*)d_in[4];
    const float* beta  = (const float*)d_in[5];
    const float* gamma = (const float*)d_in[6];
    const float* lbeta = (const float*)d_in[7];
    float* out = (float*)d_out;

    const int M = MSR_B * MSR_L;   // 16384
    const int N = MSR_D;           // 1024
    const int K = MSR_D;           // 1024
    const size_t mat  = (size_t)M * N;   // 16.78M elems
    const size_t wmat = (size_t)N * K;   // 1.05M elems

    float*    q_buf = (float*)d_ws;                              // 64 MB
    float*    v_buf = q_buf + mat;                               // 64 MB
    float*    g_buf = v_buf + mat;                               // 64 MB
    float*    carry = g_buf + mat;                               // 256 KB
    float*    sinit = carry + (size_t)MSR_NCHUNK * MSR_CH;       // 256 KB
    _Float16* xh    = (_Float16*)(sinit + (size_t)MSR_NCHUNK * MSR_CH); // 32 MB
    _Float16* wqh   = xh + mat;                                  // 2 MB each
    _Float16* wvh   = wqh + wmat;
    _Float16* wgh   = wvh + wmat;
    _Float16* woh   = wgh + wmat;
    _Float16* yh    = xh;  // x no longer needed after projections; reuse

    // ---- pack fp32 -> f16 operands
    msr_pack_f16<<<(int)(mat / 4 / 256), 256, 0, stream>>>(x,  xh,  (int)(mat / 4));
    msr_pack_f16<<<(int)(wmat / 4 / 256), 256, 0, stream>>>(Wq, wqh, (int)(wmat / 4));
    msr_pack_f16<<<(int)(wmat / 4 / 256), 256, 0, stream>>>(Wv, wvh, (int)(wmat / 4));
    msr_pack_f16<<<(int)(wmat / 4 / 256), 256, 0, stream>>>(Wg, wgh, (int)(wmat / 4));
    msr_pack_f16<<<(int)(wmat / 4 / 256), 256, 0, stream>>>(Wo, woh, (int)(wmat / 4));

    dim3 ggrid(M / GEMM_BM, N / GEMM_BN);    // 128 x 8

    // ---- projections: q, v, g
    msr_gemm_f16<<<ggrid, 256, 0, stream>>>(xh, wqh, q_buf, M, N, K);
    msr_gemm_f16<<<ggrid, 256, 0, stream>>>(xh, wvh, v_buf, M, N, K);
    msr_gemm_f16<<<ggrid, 256, 0, stream>>>(xh, wgh, g_buf, M, N, K);

    // ---- chunked retention scan; y = q*s written in-place over q_buf
    msr_scan_local<<<(MSR_CH * MSR_NCHUNK) / 256, 256, 0, stream>>>(v_buf, beta, carry);
    msr_scan_chunks<<<MSR_CH / 256, 256, 0, stream>>>(carry, beta, sinit);
    msr_scan_apply<<<(MSR_CH * MSR_NCHUNK) / 256, 256, 0, stream>>>(v_buf, q_buf, beta, sinit);

    // ---- LayerNorm + SiLU gate -> f16
    msr_ln_gate<<<M, 256, 0, stream>>>(q_buf, g_buf, gamma, lbeta, yh);

    // ---- output projection
    msr_gemm_f16<<<ggrid, 256, 0, stream>>>(yh, woh, out, M, N, K);
}
